// Decoder_33930241638561
// MI455X (gfx1250) — matrix operand order repaired and checked by reading, not compiled, not run on hardware
//
#include <hip/hip_runtime.h>

// ---------------------------------------------------------------------------
// Transformer decoder (B=32, S=200, D=1024, H=16, HD=64, L=6, FF=2048) for
// gfx1250. All matmuls run through v_wmma_f32_16x16x32_bf16 (bf16 in, fp32
// accumulate); residual stream stays fp32. GEMM A-tiles are fetched with the
// Tensor Data Mover (tensor_load_to_lds, TENSORcnt), B-tiles are staged
// transposed in LDS by the waves.
// ---------------------------------------------------------------------------

typedef unsigned short u16;
typedef __attribute__((ext_vector_type(16))) __bf16 v16bf;
typedef __attribute__((ext_vector_type(8)))  float  v8f;
typedef __attribute__((ext_vector_type(4)))  unsigned int v4u;
typedef __attribute__((ext_vector_type(8)))  int v8i;
typedef __attribute__((ext_vector_type(4)))  int v4i;

#define Bh    32
#define Sh    200
#define Dh    1024
#define Hh    16
#define HDh   64
#define Lh    6
#define FFh   2048
#define NTOK  (Bh * Sh)      // 6400
#define BHh   (Bh * Hh)      // 512
#define SKP   224            // padded key length (multiple of 32)

#if __has_builtin(__builtin_amdgcn_tensor_load_to_lds) && \
    __has_builtin(__builtin_amdgcn_s_wait_tensorcnt)
#define USE_TDM 1
#else
#define USE_TDM 0
#endif

union Frag {
    uint4 u[2];
    v16bf v;
    u16   s[16];
};

__device__ __forceinline__ u16 f2bf(float x) {
    unsigned u = __float_as_uint(x);
    unsigned r = u + 0x7FFFu + ((u >> 16) & 1u);
    return (u16)(r >> 16);
}

__device__ __forceinline__ v8f vzero8() {
    v8f z = {0.f, 0.f, 0.f, 0.f, 0.f, 0.f, 0.f, 0.f};
    return z;
}

// ---------------------------------------------------------------------------
// fp32 -> bf16 elementwise convert
// ---------------------------------------------------------------------------
__global__ void cvt_kernel(const float* __restrict__ src, u16* __restrict__ dst,
                           size_t n) {
    size_t i = (size_t)blockIdx.x * 256 + threadIdx.x;
    if (i < n) dst[i] = f2bf(src[i]);
}

// ---------------------------------------------------------------------------
// Embedding gather + sinusoidal positional encoding
// ---------------------------------------------------------------------------
__global__ void embed_kernel(const int* __restrict__ query,
                             const float* __restrict__ emb,
                             float* __restrict__ X) {
    size_t idx = (size_t)blockIdx.x * 256 + threadIdx.x;
    if (idx >= (size_t)NTOK * Dh) return;
    int m = (int)(idx >> 10);
    int d = (int)(idx & 1023);
    int s = m % Sh;
    int tok = query[m];
    float e = emb[(size_t)tok * Dh + d];
    int i2 = d & ~1;
    float div = __powf(10000.0f, (float)i2 * (1.0f / (float)Dh));
    float ang = (float)s / div;
    float pe = (d & 1) ? __cosf(ang) : __sinf(ang);
    X[idx] = e + pe;
}

// ---------------------------------------------------------------------------
// LayerNorm over D=1024; writes bf16 (for WMMA GEMM input) and/or fp32
// ---------------------------------------------------------------------------
__global__ __launch_bounds__(256)
void ln_kernel(const float* __restrict__ X, const float* __restrict__ g,
               const float* __restrict__ bta, u16* __restrict__ outb,
               float* __restrict__ outf) {
    __shared__ float red[256];
    int row = blockIdx.x;
    int t = threadIdx.x;
    const float4* xr = (const float4*)(X + (size_t)row * Dh);
    float4 v = xr[t];
    red[t] = v.x + v.y + v.z + v.w;
    __syncthreads();
    for (int st = 128; st > 0; st >>= 1) {
        if (t < st) red[t] += red[t + st];
        __syncthreads();
    }
    float mean = red[0] * (1.0f / (float)Dh);
    __syncthreads();
    float dx = v.x - mean, dy = v.y - mean, dz = v.z - mean, dw = v.w - mean;
    red[t] = dx * dx + dy * dy + dz * dz + dw * dw;
    __syncthreads();
    for (int st = 128; st > 0; st >>= 1) {
        if (t < st) red[t] += red[t + st];
        __syncthreads();
    }
    float var = red[0] * (1.0f / (float)Dh);
    float rstd = rsqrtf(var + 1e-9f);
    int d = t * 4;
    float y0 = dx * rstd * g[d + 0] + bta[d + 0];
    float y1 = dy * rstd * g[d + 1] + bta[d + 1];
    float y2 = dz * rstd * g[d + 2] + bta[d + 2];
    float y3 = dw * rstd * g[d + 3] + bta[d + 3];
    if (outb) {
        u16* o = outb + (size_t)row * Dh + d;
        o[0] = f2bf(y0); o[1] = f2bf(y1); o[2] = f2bf(y2); o[3] = f2bf(y3);
    }
    if (outf) {
        float4 o4 = {y0, y1, y2, y3};
        *(float4*)(outf + (size_t)row * Dh + d) = o4;
    }
}

// ---------------------------------------------------------------------------
// Tiled WMMA GEMM: C[M x Nc] = A_bf16[M x K] @ W_bf16[K x Nc] (+ bias)
// Block tile 128x64, 8 waves, each wave owns 2x2 16x16 accumulators.
// A tile (128x32) is DMA'd into LDS by the Tensor Data Mover: each wave issues
// one 2-D descriptor for its 16-row chunk (tile 32x16, stride K, bf16).
// B tile is staged transposed (Bt[n][k]) so fragments are 2x ds_load_b128.
// mode 0: split-heads bf16 scatter (q/k/v)
// mode 1: bf16 row-major (+optional relu)   (FF1)
// mode 2: fp32 residual accumulate into outf (Wo / W2)
// ---------------------------------------------------------------------------
__global__ __launch_bounds__(256)
void gemm_kernel(const u16* __restrict__ A, const u16* __restrict__ W,
                 const float* __restrict__ bias, int K, int Nc, int mode,
                 int relu, float* __restrict__ outf, u16* __restrict__ outb) {
    __shared__ __align__(16) u16 As[128 * 32];
    __shared__ __align__(16) u16 Bt[64 * 32];   // transposed: Bt[n][k]

    int t = threadIdx.x;
    int lane = t & 31;
    int wave = t >> 5;
    int wm = wave >> 1;          // 0..3  (32-row slice)
    int wn = wave & 1;           // 0..1  (32-col slice)
    int half = lane >> 4;
    int l16 = lane & 15;
    int m_base = blockIdx.y * 128;
    int n_base = blockIdx.x * 64;

#if USE_TDM
    // TDM descriptor pieces that are loop-invariant (per-wave 16-row chunk).
    unsigned lds_base = (unsigned)(size_t)(void*)As + (unsigned)(wave * 16 * 32 * 2);
    unsigned long long ga_row0 =
        (unsigned long long)(size_t)(const void*)(A + (size_t)(m_base + wave * 16) * K);
    // D# group 1: wg_mask=0, data_size=1 (2B), no barrier/iterate/pad,
    // tensor_dim0=K, tensor_dim1=NTOK, tile_dim0=32, tile_dim1=16,
    // tile_dim2=0, tensor_dim0_stride=K, tensor_dim1_stride=0.
    int g1w0 = 0x10000;
    int g1w1 = (int)(((unsigned)K & 0xFFFFu) << 16);
    int g1w2 = (int)(((unsigned)K >> 16) | (((unsigned)NTOK & 0xFFFFu) << 16));
    int g1w3 = (int)(((unsigned)NTOK >> 16) | (32u << 16));
    int g1w4 = 16;
    int g1w5 = K;                 // tensor_dim0_stride[31:0]
    v8i g1 = {g1w0, g1w1, g1w2, g1w3, g1w4, g1w5, 0, 0};
    v4i gz4 = {0, 0, 0, 0};
    v8i gz8 = {0, 0, 0, 0, 0, 0, 0, 0};
#endif

    v8f acc[2][2];
    acc[0][0] = vzero8(); acc[0][1] = vzero8();
    acc[1][0] = vzero8(); acc[1][1] = vzero8();

    for (int k0 = 0; k0 < K; k0 += 32) {
#if USE_TDM
        {   // A tile via Tensor Data Mover (one op per wave, EXEC ignored)
            unsigned long long ga = ga_row0 + (unsigned long long)(k0 * 2);
            v4u g0 = {1u,                                   // count=1
                      lds_base,                             // lds_addr
                      (unsigned)(ga & 0xFFFFFFFFull),       // global_addr[31:0]
                      (unsigned)((ga >> 32) & 0x1FFFFFFu) | (2u << 30)}; // +type=2
            __builtin_amdgcn_tensor_load_to_lds(g0, g1, gz4, gz4, gz8, 0);
        }
#else
#pragma unroll
        for (int it = 0; it < 2; ++it) {
            int idx = t + it * 256;
            int row = idx >> 2;
            int c = (idx & 3) * 8;
            uint4 val = *(const uint4*)(A + (size_t)(m_base + row) * K + k0 + c);
            *(uint4*)(As + row * 32 + c) = val;
        }
#endif
        // cooperative B tile load, stored transposed Bt[n][k]
        {
            int k = t >> 3;
            int nc = (t & 7) * 8;
            union { uint4 u; u16 s[8]; } tv;
            tv.u = *(const uint4*)(W + (size_t)(k0 + k) * Nc + n_base + nc);
#pragma unroll
            for (int j = 0; j < 8; ++j) Bt[(nc + j) * 32 + k] = tv.s[j];
        }
        if (k0 + 32 < K) {
            __builtin_prefetch(W + (size_t)(k0 + 32 + (t >> 3)) * Nc + n_base, 0, 1);
        }
#if USE_TDM
        __builtin_amdgcn_s_wait_tensorcnt(0);
#endif
        __syncthreads();

        Frag a[2], b[2];
#pragma unroll
        for (int mt = 0; mt < 2; ++mt) {
            int row = wm * 32 + mt * 16 + l16;
            const u16* p = As + row * 32 + half * 8;
            a[mt].u[0] = *(const uint4*)p;          // K = half*8 .. +7
            a[mt].u[1] = *(const uint4*)(p + 16);   // K = 16+half*8 .. +7
        }
#pragma unroll
        for (int nt = 0; nt < 2; ++nt) {
            int col = wn * 32 + nt * 16 + l16;
            const u16* p = Bt + col * 32 + half * 8;
            b[nt].u[0] = *(const uint4*)p;          // K = half*8 .. +7
            b[nt].u[1] = *(const uint4*)(p + 16);   // K = 16+half*8 .. +7
        }
#pragma unroll
        for (int mt = 0; mt < 2; ++mt)
#pragma unroll
            for (int nt = 0; nt < 2; ++nt)
                acc[mt][nt] = __builtin_amdgcn_wmma_f32_16x16x32_bf16(
                    false, a[mt].v, false, b[nt].v, (short)0, acc[mt][nt],
                    false, false);
        __syncthreads();
    }

    // epilogue: C layout row = half*8 + r, col = lane%16
#pragma unroll
    for (int mt = 0; mt < 2; ++mt)
#pragma unroll
        for (int nt = 0; nt < 2; ++nt)
#pragma unroll
            for (int r = 0; r < 8; ++r) {
                int m = m_base + wm * 32 + mt * 16 + half * 8 + r;
                int n = n_base + wn * 32 + nt * 16 + l16;
                float v = acc[mt][nt][r];
                if (bias) v += bias[n];
                if (relu) v = fmaxf(v, 0.0f);
                if (mode == 0) {
                    int h = n >> 6, hd = n & 63;
                    int bidx = m / Sh, s = m - bidx * Sh;
                    outb[(((size_t)bidx * Hh + h) * Sh + s) * HDh + hd] = f2bf(v);
                } else if (mode == 1) {
                    outb[(size_t)m * Nc + n] = f2bf(v);
                } else {
                    outf[(size_t)m * Nc + n] += v;
                }
            }
}

// ---------------------------------------------------------------------------
// Attention scores: S[bh,q,k] = sum_d Q[bh,q,d] * K[bh,k,d]  (WMMA, HD=64)
// One wave computes one 16x16 score tile; fragments load directly from global.
// ---------------------------------------------------------------------------
__global__ __launch_bounds__(32)
void attn_scores_kernel(const u16* __restrict__ Q, const u16* __restrict__ Kb,
                        float* __restrict__ Sc) {
    int bh = blockIdx.z;
    int q0 = blockIdx.y * 16;
    int k0 = blockIdx.x * 16;
    int lane = threadIdx.x & 31;
    int half = lane >> 4;
    int l16 = lane & 15;
    const u16* Qr = Q + (size_t)bh * Sh * HDh;
    const u16* Kr = Kb + (size_t)bh * Sh * HDh;
    int q = q0 + l16;  int qc = (q < Sh) ? q : (Sh - 1);
    int kc0 = k0 + l16; int kc = (kc0 < Sh) ? kc0 : (Sh - 1);

    v8f acc = vzero8();
#pragma unroll
    for (int ds = 0; ds < HDh; ds += 32) {
        Frag a, b;
        const u16* ap = Qr + (size_t)qc * HDh + ds + half * 8;
        a.u[0] = *(const uint4*)ap;
        a.u[1] = *(const uint4*)(ap + 16);
        const u16* bp = Kr + (size_t)kc * HDh + ds + half * 8;
        b.u[0] = *(const uint4*)bp;
        b.u[1] = *(const uint4*)(bp + 16);
        acc = __builtin_amdgcn_wmma_f32_16x16x32_bf16(false, a.v, false, b.v,
                                                      (short)0, acc, false,
                                                      false);
    }
    float* out = Sc + (size_t)bh * Sh * Sh;
#pragma unroll
    for (int r = 0; r < 8; ++r) {
        int qq = q0 + half * 8 + r;
        int kk = k0 + l16;
        if (qq < Sh && kk < Sh) out[(size_t)qq * Sh + kk] = acc[r];
    }
}

// ---------------------------------------------------------------------------
// Softmax over 200 keys (1 wave per row), mask = causal | pad, scale 1/8.
// Writes bf16 probs padded with zeros to SKP=224.
// ---------------------------------------------------------------------------
__global__ __launch_bounds__(32)
void softmax_kernel(const float* __restrict__ Sc, u16* __restrict__ P,
                    const int* __restrict__ query, int self_mask) {
    int q = blockIdx.x;
    int bh = blockIdx.y;
    int b = bh >> 4;
    int lane = threadIdx.x;
    const float* row = Sc + ((size_t)bh * Sh + q) * Sh;
    float vals[7];
    float mx = -3.4e38f;
#pragma unroll
    for (int i = 0; i < 7; ++i) {
        int k = lane + i * 32;
        float v = -3.4e38f;
        if (k < Sh) {
            v = row[k] * 0.125f;  // 1/sqrt(64)
            if (self_mask && ((k > q) || (query[(size_t)b * Sh + k] == 0)))
                v = -3.4e38f;
        }
        vals[i] = v;
        mx = fmaxf(mx, v);
    }
    for (int off = 16; off; off >>= 1) mx = fmaxf(mx, __shfl_xor(mx, off, 32));
    float sum = 0.f;
#pragma unroll
    for (int i = 0; i < 7; ++i) {
        int k = lane + i * 32;
        float e = (k < Sh) ? __expf(vals[i] - mx) : 0.f;
        vals[i] = e;
        sum += e;
    }
    for (int off = 16; off; off >>= 1) sum += __shfl_xor(sum, off, 32);
    float inv = 1.0f / sum;
    u16* pr = P + ((size_t)bh * Sh + q) * SKP;
#pragma unroll
    for (int i = 0; i < 7; ++i) {
        int k = lane + i * 32;
        if (k < Sh) pr[k] = f2bf(vals[i] * inv);
    }
    if (lane < SKP - Sh) pr[Sh + lane] = 0;  // zero padding for WMMA reads
}

// ---------------------------------------------------------------------------
// Context: CTX[b,s,h*64+hd] = sum_k P[bh,s,k] * V[bh,k,hd]  (WMMA over k)
// Block = 2 waves; wave wv covers 32 of the 64 head dims.
// ---------------------------------------------------------------------------
__global__ __launch_bounds__(64)
void attn_ctx_kernel(const u16* __restrict__ P, const u16* __restrict__ V,
                     u16* __restrict__ CTX) {
    int bh = blockIdx.y;
    int q0 = blockIdx.x * 16;
    int lane = threadIdx.x & 31;
    int wv = threadIdx.x >> 5;  // 0..1
    int half = lane >> 4;
    int l16 = lane & 15;
    const u16* Pr = P + (size_t)bh * Sh * SKP;
    const u16* Vr = V + (size_t)bh * Sh * HDh;
    int q = q0 + l16;
    int qc = (q < Sh) ? q : (Sh - 1);

    v8f acc[2];
    acc[0] = vzero8();
    acc[1] = vzero8();
    for (int ks = 0; ks < SKP; ks += 32) {
        Frag a;
        const u16* ap = Pr + (size_t)qc * SKP + ks + half * 8;
        a.u[0] = *(const uint4*)ap;
        a.u[1] = *(const uint4*)(ap + 16);
#pragma unroll
        for (int nt = 0; nt < 2; ++nt) {
            int col = wv * 32 + nt * 16 + l16;
            Frag b;
#pragma unroll
            for (int e = 0; e < 16; ++e) {
                int kk = ks + half * 8 + (e < 8 ? e : 8 + e);
                b.s[e] = (kk < Sh) ? Vr[(size_t)kk * HDh + col] : (u16)0;
            }
            acc[nt] = __builtin_amdgcn_wmma_f32_16x16x32_bf16(
                false, a.v, false, b.v, (short)0, acc[nt], false, false);
        }
    }
    int bb = bh >> 4, hh = bh & 15;
#pragma unroll
    for (int nt = 0; nt < 2; ++nt)
#pragma unroll
        for (int r = 0; r < 8; ++r) {
            int qq = q0 + half * 8 + r;
            if (qq < Sh) {
                int nn = wv * 32 + nt * 16 + l16;
                CTX[((size_t)(bb * Sh + qq)) * Dh + hh * HDh + nn] =
                    f2bf(acc[nt][r]);
            }
        }
}

// ---------------------------------------------------------------------------
// Host-side orchestration
// ---------------------------------------------------------------------------
extern "C" void kernel_launch(void* const* d_in, const int* in_sizes, int n_in,
                              void* d_out, int out_size, void* d_ws,
                              size_t ws_size, hipStream_t stream) {
    (void)in_sizes; (void)n_in; (void)out_size; (void)ws_size;

    // inputs (setup_inputs dict order)
    const int*   query = (const int*)  d_in[0];
    const float* key   = (const float*)d_in[1];
    const float* value = (const float*)d_in[2];
    const float* emb   = (const float*)d_in[3];
    const float* ln1_g = (const float*)d_in[4];
    const float* ln2_g = (const float*)d_in[5];
    const float* ln3_g = (const float*)d_in[6];
    const float* ln1_b = (const float*)d_in[7];
    const float* ln2_b = (const float*)d_in[8];
    const float* ln3_b = (const float*)d_in[9];
    const float* Wq_s  = (const float*)d_in[10];
    const float* Wk_s  = (const float*)d_in[11];
    const float* Wv_s  = (const float*)d_in[12];
    const float* Wo_s  = (const float*)d_in[13];
    const float* Wq_c  = (const float*)d_in[14];
    const float* Wk_c  = (const float*)d_in[15];
    const float* Wv_c  = (const float*)d_in[16];
    const float* Wo_c  = (const float*)d_in[17];
    const float* bq_s  = (const float*)d_in[18];
    const float* bk_s  = (const float*)d_in[19];
    const float* bv_s  = (const float*)d_in[20];
    const float* bq_c  = (const float*)d_in[21];
    const float* bk_c  = (const float*)d_in[22];
    const float* bv_c  = (const float*)d_in[23];
    const float* W1    = (const float*)d_in[24];
    const float* bf1   = (const float*)d_in[25];
    const float* W2    = (const float*)d_in[26];
    const float* bf2   = (const float*)d_in[27];
    const float* lnf_g = (const float*)d_in[28];
    const float* lnf_b = (const float*)d_in[29];

    // workspace carving
    char* wsb = (char*)d_ws;
    size_t off = 0;
    auto take = [&](size_t bytes) -> char* {
        char* p = wsb + off;
        off = (off + bytes + 255) & ~(size_t)255;
        return p;
    };
    const size_t szDD = (size_t)Lh * Dh * Dh;     // elems per [L,D,D]
    const size_t szDF = (size_t)Lh * Dh * FFh;    // elems per [L,D,FF]

    u16* WQS = (u16*)take(szDD * 2);
    u16* WKS = (u16*)take(szDD * 2);
    u16* WVS = (u16*)take(szDD * 2);
    u16* WOS = (u16*)take(szDD * 2);
    u16* WQC = (u16*)take(szDD * 2);
    u16* WKC = (u16*)take(szDD * 2);
    u16* WVC = (u16*)take(szDD * 2);
    u16* WOC = (u16*)take(szDD * 2);
    u16* W1B = (u16*)take(szDF * 2);
    u16* W2B = (u16*)take(szDF * 2);
    u16* KEYB = (u16*)take((size_t)NTOK * Dh * 2);
    u16* VALB = (u16*)take((size_t)NTOK * Dh * 2);
    float* Xr = (float*)take((size_t)NTOK * Dh * 4);
    u16* XN  = (u16*)take((size_t)NTOK * Dh * 2);
    u16* Qb  = (u16*)take((size_t)BHh * Sh * HDh * 2);
    u16* Kb  = (u16*)take((size_t)BHh * Sh * HDh * 2);
    u16* Vb  = (u16*)take((size_t)BHh * Sh * HDh * 2);
    float* Sc = (float*)take((size_t)BHh * Sh * Sh * 4);
    u16* Pb  = (u16*)take((size_t)BHh * Sh * SKP * 2);
    u16* CTX = (u16*)take((size_t)NTOK * Dh * 2);
    u16* FF1 = (u16*)take((size_t)NTOK * FFh * 2);

    auto cvt = [&](const float* s, u16* d, size_t n) {
        int blocks = (int)((n + 255) / 256);
        cvt_kernel<<<blocks, 256, 0, stream>>>(s, d, n);
    };
    // one-time (per launch) weight / KV conversion to bf16
    cvt(Wq_s, WQS, szDD); cvt(Wk_s, WKS, szDD);
    cvt(Wv_s, WVS, szDD); cvt(Wo_s, WOS, szDD);
    cvt(Wq_c, WQC, szDD); cvt(Wk_c, WKC, szDD);
    cvt(Wv_c, WVC, szDD); cvt(Wo_c, WOC, szDD);
    cvt(W1, W1B, szDF);   cvt(W2, W2B, szDF);
    cvt(key,   KEYB, (size_t)NTOK * Dh);
    cvt(value, VALB, (size_t)NTOK * Dh);

    embed_kernel<<<(NTOK * Dh + 255) / 256, 256, 0, stream>>>(query, emb, Xr);

    auto gemm = [&](const u16* A, const u16* W, const float* bias, int K,
                    int Nc, int mode, int relu, float* of, u16* ob) {
        dim3 g(Nc / 64, NTOK / 128);
        gemm_kernel<<<g, 256, 0, stream>>>(A, W, bias, K, Nc, mode, relu, of, ob);
    };
    auto attention = [&](int self) {
        attn_scores_kernel<<<dim3(13, 13, BHh), 32, 0, stream>>>(Qb, Kb, Sc);
        softmax_kernel<<<dim3(Sh, BHh), 32, 0, stream>>>(Sc, Pb, query, self);
        attn_ctx_kernel<<<dim3(13, BHh), 64, 0, stream>>>(Pb, Vb, CTX);
    };

    for (int l = 0; l < Lh; ++l) {
        size_t oDD = (size_t)l * Dh * Dh;
        size_t oDF = (size_t)l * Dh * FFh;
        size_t oD = (size_t)l * Dh;
        size_t oF = (size_t)l * FFh;

        // --- self attention ---
        ln_kernel<<<NTOK, 256, 0, stream>>>(Xr, ln1_g + oD, ln1_b + oD, XN, nullptr);
        gemm(XN, WQS + oDD, bq_s + oD, Dh, Dh, 0, 0, nullptr, Qb);
        gemm(XN, WKS + oDD, bk_s + oD, Dh, Dh, 0, 0, nullptr, Kb);
        gemm(XN, WVS + oDD, bv_s + oD, Dh, Dh, 0, 0, nullptr, Vb);
        attention(1);
        gemm(CTX, WOS + oDD, nullptr, Dh, Dh, 2, 0, Xr, nullptr);

        // --- cross attention ---
        ln_kernel<<<NTOK, 256, 0, stream>>>(Xr, ln2_g + oD, ln2_b + oD, XN, nullptr);
        gemm(XN,   WQC + oDD, bq_c + oD, Dh, Dh, 0, 0, nullptr, Qb);
        gemm(KEYB, WKC + oDD, bk_c + oD, Dh, Dh, 0, 0, nullptr, Kb);
        gemm(VALB, WVC + oDD, bv_c + oD, Dh, Dh, 0, 0, nullptr, Vb);
        attention(0);
        gemm(CTX, WOC + oDD, nullptr, Dh, Dh, 2, 0, Xr, nullptr);

        // --- feed forward ---
        ln_kernel<<<NTOK, 256, 0, stream>>>(Xr, ln3_g + oD, ln3_b + oD, XN, nullptr);
        gemm(XN, W1B + oDF, bf1 + oF, Dh, FFh, 1, 1, nullptr, FF1);
        gemm(FF1, W2B + oDF, bf2 + oD, FFh, Dh, 2, 0, Xr, nullptr);
    }

    // final layernorm -> fp32 output [B,S,D]
    ln_kernel<<<NTOK, 256, 0, stream>>>(Xr, lnf_g, lnf_b, nullptr, (float*)d_out);
}
